// BN_ReLU_DConv_412316860616
// MI455X (gfx1250) — matrix operand order repaired and checked by reading, not compiled, not run on hardware
//
#include <hip/hip_runtime.h>
#include <hip/hip_bf16.h>

// ---------------------------------------------------------------------------
// BN+ReLU -> offset conv -> deformable conv, with the big contraction
// (M=32768, N=128, K=1152) on the CDNA5 WMMA pipe (v_wmma_f32_16x16x32_f16).
// GEMM A-tile is staged once per workgroup into LDS (bank-conflict-padded)
// and shared by all 8 waves; B fragments stream from global (L2-resident).
// Workspace layout (bytes):
//   [0,              16777216)  h       : f32 (8,128,64,64)   BN+ReLU output
//   [16777216,       19136512)  offsets : f32 (8,18,64,64)
//   [19136512,       94633984)  A       : f16 [32768][1152]   sampled values
//   [94633984,       94928896)  Bw      : f16 [128][1152]     w_dconv cast
// Requires ws_size >= ~95 MB.
// ---------------------------------------------------------------------------

typedef __attribute__((ext_vector_type(16))) _Float16 v16h;
typedef __attribute__((ext_vector_type(8)))  _Float16 h8;
typedef __attribute__((ext_vector_type(8)))  float    v8f;
typedef __attribute__((ext_vector_type(4)))  float    v4f_;

#define BB   8
#define CC   128
#define OO   128
#define HH   64
#define WW   64
#define HW   4096          // 64*64
#define KK9  9
#define KDIM 1152          // 128*9
#define MTOT 32768         // 8*64*64
#define LDSROW 1160        // 1152 + 8 pad halves -> 580 dwords, 580%64=4

// ---------------- Kernel 1: BN + ReLU (elementwise) ----------------
__global__ void k_bn_relu(const float* __restrict__ x,
                          const float* __restrict__ gamma,
                          const float* __restrict__ beta,
                          const float* __restrict__ mean,
                          const float* __restrict__ var,
                          float* __restrict__ h, int n) {
    int i = blockIdx.x * blockDim.x + threadIdx.x;
    if (i >= n) return;
    int c = (i >> 12) & (CC - 1);          // (i / 4096) % 128
    float inv = gamma[c] * rsqrtf(var[c] + 1e-5f);
    float v = (x[i] - mean[c]) * inv + beta[c];
    h[i] = v > 0.0f ? v : 0.0f;
}

// ---------------- Kernel 2: 3x3 conv for offsets (18 out ch) ----------------
__global__ void k_offset_conv(const float* __restrict__ h,
                              const float* __restrict__ w_off,
                              const float* __restrict__ b_off,
                              float* __restrict__ offs) {
    int tid = blockIdx.x * blockDim.x + threadIdx.x;
    if (tid >= BB * 18 * HW) return;
    int pix = tid & (HW - 1);
    int t   = tid >> 12;
    int j   = t % 18;
    int b   = t / 18;
    int hy  = pix >> 6, wx = pix & 63;

    float acc = b_off[j];
    const float* hb = h + (size_t)b * CC * HW;
    const float* wj = w_off + (size_t)j * CC * 9;
    for (int c = 0; c < CC; ++c) {
        const float* hc = hb + (size_t)c * HW;
        const float* wc = wj + c * 9;
        #pragma unroll
        for (int kh = 0; kh < 3; ++kh) {
            int y = hy + kh - 1;
            if (y < 0 || y >= HH) continue;
            #pragma unroll
            for (int kw = 0; kw < 3; ++kw) {
                int xx = wx + kw - 1;
                if (xx < 0 || xx >= WW) continue;
                acc += hc[y * WW + xx] * wc[kh * 3 + kw];
            }
        }
    }
    offs[tid] = acc;
}

// ---------------- Kernel 3: bilinear sampling -> A[M][KDIM] f16 -------------
// A[m][c*9+kk] = bilinear(h[b,c], py, px) with reference semantics
__global__ void k_sample(const float* __restrict__ h,
                         const float* __restrict__ offs,
                         _Float16* __restrict__ A) {
    int tid = blockIdx.x * blockDim.x + threadIdx.x;
    if (tid >= MTOT * KK9) return;
    int m  = tid / KK9;
    int kk = tid - m * KK9;
    int b   = m >> 12;
    int pix = m & (HW - 1);
    int hy  = pix >> 6, wx = pix & 63;

    float dy = offs[((size_t)(b * 18 + kk * 2 + 0) << 12) + pix];
    float dx = offs[((size_t)(b * 18 + kk * 2 + 1) << 12) + pix];
    float py = dy + (float)hy + (float)(kk / 3 - 1);
    float px = dx + (float)wx + (float)(kk % 3 - 1);
    float y0f = floorf(py), x0f = floorf(px);
    int   y0  = (int)y0f,   x0  = (int)x0f;
    float wy = py - y0f, wxf = px - x0f;

    float w00 = (1.f - wy) * (1.f - wxf);
    float w01 = (1.f - wy) * wxf;
    float w10 = wy * (1.f - wxf);
    float w11 = wy * wxf;

    int y1 = y0 + 1, x1 = x0 + 1;
    bool vy0 = (y0 >= 0) & (y0 < HH), vy1 = (y1 >= 0) & (y1 < HH);
    bool vx0 = (x0 >= 0) & (x0 < WW), vx1 = (x1 >= 0) & (x1 < WW);
    int cy0 = min(max(y0, 0), HH - 1), cy1 = min(max(y1, 0), HH - 1);
    int cx0 = min(max(x0, 0), WW - 1), cx1 = min(max(x1, 0), WW - 1);
    int i00 = cy0 * WW + cx0, i01 = cy0 * WW + cx1;
    int i10 = cy1 * WW + cx0, i11 = cy1 * WW + cx1;
    bool v00 = vy0 & vx0, v01 = vy0 & vx1, v10 = vy1 & vx0, v11 = vy1 & vx1;

    const float* hb = h + ((size_t)b * CC << 12);
    _Float16* arow = A + (size_t)m * KDIM + kk;
    for (int c = 0; c < CC; ++c) {
        const float* hc = hb + ((size_t)c << 12);
        float val = w00 * (v00 ? hc[i00] : 0.f)
                  + w01 * (v01 ? hc[i01] : 0.f)
                  + w10 * (v10 ? hc[i10] : 0.f)
                  + w11 * (v11 ? hc[i11] : 0.f);
        arow[c * KK9] = (_Float16)val;
    }
}

// ---------------- Kernel 4: cast weights to f16 [O][KDIM] ----------------
// w_dconv is (O,C,3,3) row-major == [O][KDIM] with kdim = c*9 + kk already.
__global__ void k_pack_w(const float* __restrict__ w, _Float16* __restrict__ Bw,
                         int n) {
    int i = blockIdx.x * blockDim.x + threadIdx.x;
    if (i < n) Bw[i] = (_Float16)w[i];
}

// ---------------- Kernel 5: WMMA GEMM: out[M][N] = A * Bw^T + bias ----------
// Block = 256 threads = 8 waves. Block owns one 16-row M tile; the A-tile
// (16 x 1152 f16) is staged to LDS once (padded rows: lane r -> banks
// 4r..4r+3, conflict-free) and shared by all 8 waves. Wave w computes the
// 16x16 N-tile at col 16*w with 36 x v_wmma_f32_16x16x32_f16.
__global__ void __launch_bounds__(256)
k_wmma_gemm(const _Float16* __restrict__ A,
            const _Float16* __restrict__ Bw,
            const float* __restrict__ bias,
            float* __restrict__ out) {
    __shared__ _Float16 As[16 * LDSROW];     // 37,120 B of the 320 KB WGP LDS

    const int t    = threadIdx.x;
    const int lane = t & 31;
    const int wave = t >> 5;                  // 0..7 -> N tile
    const int half = lane >> 4;               // 0/1
    const int r    = lane & 15;

    const int rowBase = blockIdx.x << 4;      // M tile (16 rows, same b)
    const int colBase = wave << 4;            // N tile

    // ---- Stage A tile to LDS: 16 rows x 144 chunks of 8 halves (16 B) ----
    // 2304 chunks over 256 threads = 9 per thread.
    {
        const _Float16* ag = A + (size_t)rowBase * KDIM;
        #pragma unroll
        for (int i = 0; i < 9; ++i) {
            int ci   = t + (i << 8);          // 0..2303
            int row  = ci / 144;
            int col8 = ci - row * 144;
            *(h8*)(As + row * LDSROW + (col8 << 3)) =
                *(const h8*)(ag + (size_t)row * KDIM + (col8 << 3));
        }
    }
    __syncthreads();

    // Per-lane fragment pointers per the CDNA5 16-bit VGPR layouts:
    //  A 16x32: lane row = r; elems j0..7  at k = kb + 8*half + j,
    //                         elems j8..15 at k = kb + 16 + 8*half + (j-8)
    //  B 32x16 (stored [O][KDIM] so K is contiguous per lane):
    //           lane col = r; same k order as A
    const _Float16* alds = As + r * LDSROW + 8 * half;
    const _Float16* brow = Bw + (size_t)(colBase + r) * KDIM + 8 * half;

    union AV { v16h v; h8 h[2]; };
    v8f acc = {};

    #pragma unroll 4
    for (int kb = 0; kb < KDIM; kb += 32) {
        AV av, bv;
        av.h[0] = *(const h8*)(alds + kb);
        av.h[1] = *(const h8*)(alds + kb + 16);
        bv.h[0] = *(const h8*)(brow + kb);
        bv.h[1] = *(const h8*)(brow + kb + 16);
        acc = __builtin_amdgcn_wmma_f32_16x16x32_f16(
                false, av.v, false, bv.v, (short)0, acc, false, false);
    }

    // D layout: elem i -> row rowBase + 8*half + i (contiguous), col = r.
    // out is (B,O,H,W): addr = b*O*HW + o*HW + (m % HW); rows share b.
    const int o = colBase + r;
    const int b = rowBase >> 12;
    const float bo = bias[o];
    float* op = out + ((size_t)b * OO << 12) + ((size_t)o << 12)
                    + (rowBase & (HW - 1)) + 8 * half;
    v4f_ lo, hi;
    #pragma unroll
    for (int i = 0; i < 4; ++i) { lo[i] = acc[i] + bo; hi[i] = acc[i + 4] + bo; }
    *(v4f_*)(op)     = lo;
    *(v4f_*)(op + 4) = hi;
}

// ---------------------------------------------------------------------------
extern "C" void kernel_launch(void* const* d_in, const int* in_sizes, int n_in,
                              void* d_out, int out_size, void* d_ws, size_t ws_size,
                              hipStream_t stream) {
    const float* x        = (const float*)d_in[0];
    const float* bn_gamma = (const float*)d_in[1];
    const float* bn_beta  = (const float*)d_in[2];
    const float* bn_mean  = (const float*)d_in[3];
    const float* bn_var   = (const float*)d_in[4];
    const float* w_off    = (const float*)d_in[5];
    const float* b_off    = (const float*)d_in[6];
    const float* w_dconv  = (const float*)d_in[7];
    const float* b_dconv  = (const float*)d_in[8];
    float* out = (float*)d_out;

    char* ws = (char*)d_ws;
    float*    h    = (float*)(ws);                         // 16,777,216 B
    float*    offs = (float*)(ws + 16777216);              //  2,359,296 B
    _Float16* A    = (_Float16*)(ws + 19136512);           // 75,497,472 B
    _Float16* Bw   = (_Float16*)(ws + 94633984);           //    294,912 B

    // 1) BN + ReLU
    {
        int n = BB * CC * HW;
        k_bn_relu<<<(n + 255) / 256, 256, 0, stream>>>(
            x, bn_gamma, bn_beta, bn_mean, bn_var, h, n);
    }
    // 2) offset conv
    {
        int n = BB * 18 * HW;
        k_offset_conv<<<(n + 255) / 256, 256, 0, stream>>>(h, w_off, b_off, offs);
    }
    // 3) bilinear sampling -> A (f16)
    {
        int n = MTOT * KK9;
        k_sample<<<(n + 255) / 256, 256, 0, stream>>>(h, offs, A);
    }
    // 4) weight cast -> Bw (f16)
    {
        int n = OO * KDIM;
        k_pack_w<<<(n + 255) / 256, 256, 0, stream>>>(w_dconv, Bw, n);
    }
    // 5) WMMA GEMM + bias -> out (LDS-staged A tile shared by 8 waves)
    {
        k_wmma_gemm<<<MTOT / 16, 256, 0, stream>>>(A, Bw, b_dconv, out);
    }
}
